// MultiHeadSelfAttention_77249281786555
// MI455X (gfx1250) — matrix operand order repaired and checked by reading, not compiled, not run on hardware
//
#include <hip/hip_runtime.h>

typedef __attribute__((ext_vector_type(16))) __bf16 v16bf;
typedef __attribute__((ext_vector_type(8)))  __bf16 v8bf;
typedef __attribute__((ext_vector_type(8)))  float  v8f;

#define E_DIM 1024
#define H_DIM 16
#define D_DIM 64
#define B_DIM 4
#define S_DIM 2048
#define M_ROWS (B_DIM * S_DIM)   // 8192 token rows

// ---------- helpers ----------
__device__ __forceinline__ unsigned short f2bf_rne(float f) {
    union { float f; unsigned int u; } v; v.f = f;
    unsigned int u = v.u;
    unsigned int r = u + 0x7fffu + ((u >> 16) & 1u);   // round-to-nearest-even
    return (unsigned short)(r >> 16);
}

__device__ __forceinline__ v16bf cat8(v8bf lo, v8bf hi) {
    return __builtin_shufflevector(lo, hi, 0,1,2,3,4,5,6,7,8,9,10,11,12,13,14,15);
}

__device__ __forceinline__ v8f wmma_bf16(v16bf a, v16bf b, v8f c) {
    // (neg_a, A, neg_b, B, c_mod, C, reuse_a, reuse_b)
    return __builtin_amdgcn_wmma_f32_16x16x32_bf16(false, a, false, b, (short)0, c, false, false);
}

// ---------- phase 0: conversions ----------
__global__ void __launch_bounds__(256) convert_x_kernel(const float* __restrict__ x,
                                                        unsigned short* __restrict__ xb, int n) {
    int i = blockIdx.x * blockDim.x + threadIdx.x;
    if (i < n) xb[i] = f2bf_rne(x[i]);
}

// wt holds 4 matrices [mat][N=1024][K=1024], transposed from row-major [K][N].
// wq (mat 0) is pre-scaled by 1/sqrt(D) = 0.125 so scores come out pre-scaled.
__global__ void __launch_bounds__(256) convert_w_kernel(const float* __restrict__ wq,
                                                        const float* __restrict__ wk,
                                                        const float* __restrict__ wv,
                                                        const float* __restrict__ wo,
                                                        unsigned short* __restrict__ wt) {
    int i = blockIdx.x * blockDim.x + threadIdx.x;   // 0 .. 4M-1
    int mat = i >> 20;
    int r   = i & ((1 << 20) - 1);
    int n   = r >> 10;
    int k   = r & 1023;
    const float* w = (mat == 0) ? wq : (mat == 1) ? wk : (mat == 2) ? wv : wo;
    float s = (mat == 0) ? 0.125f : 1.0f;
    wt[i] = f2bf_rne(w[(size_t)k * E_DIM + n] * s);
}

// ---------- phase 1: fused QKV projection (bf16 WMMA GEMM, 32x64 per wave) ----------
// waves: mat(3) x mtile(256, 32 rows each) x ngroup(16); 8 waves/block.
__global__ void __launch_bounds__(256) qkv_gemm_kernel(
    const unsigned short* __restrict__ xb,
    const unsigned short* __restrict__ wt,
    const float* __restrict__ bq, const float* __restrict__ bk, const float* __restrict__ bv,
    unsigned short* __restrict__ Qb,   // [B,H,S,D] bf16 (pre-scaled)
    unsigned short* __restrict__ Kb,   // [B,H,S,D] bf16
    unsigned short* __restrict__ VT)   // [B,H,D,S] bf16 (transposed V)
{
    int wave  = blockIdx.x * (blockDim.x >> 5) + (threadIdx.x >> 5);
    int lane  = threadIdx.x & 31;
    int half  = lane >> 4;
    int l16   = lane & 15;

    int ng    = wave & 15;
    int mtile = (wave >> 4) & 255;
    int mat   = wave >> 12;                 // 0=Q, 1=K, 2=V

    const unsigned short* W = wt + ((size_t)mat << 20);
    int m0 = mtile * 32;
    const unsigned short* arow0 = xb + (size_t)(m0 + l16) * E_DIM;
    const unsigned short* arow1 = xb + (size_t)(m0 + 16 + l16) * E_DIM;
    const unsigned short* brow0 = W + (size_t)(ng * 64 + l16) * E_DIM + half * 16;

    v8f acc[2][4] = {};
    for (int k0 = 0; k0 < E_DIM; k0 += 32) {
        v16bf a0 = cat8(*(const v8bf*)(arow0 + k0 + half * 16),
                        *(const v8bf*)(arow0 + k0 + half * 16 + 8));
        v16bf a1 = cat8(*(const v8bf*)(arow1 + k0 + half * 16),
                        *(const v8bf*)(arow1 + k0 + half * 16 + 8));
        __builtin_prefetch(arow0 + k0 + 32 + half * 8, 0, 1);
        __builtin_prefetch(arow1 + k0 + 32 + half * 8, 0, 1);
#pragma unroll
        for (int j = 0; j < 4; ++j) {
            v16bf b = *(const v16bf*)(brow0 + (size_t)(j * 16) * E_DIM + k0);
            acc[0][j] = wmma_bf16(a0, b, acc[0][j]);
            acc[1][j] = wmma_bf16(a1, b, acc[1][j]);
        }
    }

    const float* bias = (mat == 0) ? bq : (mat == 1) ? bk : bv;
    float bsc = (mat == 0) ? 0.125f : 1.0f;
#pragma unroll
    for (int j = 0; j < 4; ++j) {
        int e = ng * 64 + j * 16 + l16;
        int h = e >> 6, d = e & 63;
        float bval = bias[e] * bsc;
#pragma unroll
        for (int g = 0; g < 2; ++g) {
#pragma unroll
            for (int r = 0; r < 8; ++r) {
                int row = m0 + g * 16 + half * 8 + r;
                int b   = row >> 11;            // / S_DIM
                int s   = row & (S_DIM - 1);
                unsigned short o = f2bf_rne(acc[g][j][r] + bval);
                size_t bh = (size_t)(b * H_DIM + h);
                if (mat == 0)      Qb[(bh * S_DIM + s) * D_DIM + d] = o;
                else if (mat == 1) Kb[(bh * S_DIM + s) * D_DIM + d] = o;
                else               VT[(bh * D_DIM + d) * S_DIM + s] = o;
            }
        }
    }
}

// ---------- phase 2: flash attention (one wave per (b,h,16-query block)) ----------
__global__ void __launch_bounds__(256) attention_kernel(
    const unsigned short* __restrict__ Qb,
    const unsigned short* __restrict__ Kb,
    const unsigned short* __restrict__ VT,
    unsigned short* __restrict__ Ob)     // [B,S,E] bf16
{
    __shared__ __align__(32) unsigned short pbuf[8][16 * 32];  // per-wave P tile
    int wslot = threadIdx.x >> 5;
    int wave  = blockIdx.x * 8 + wslot;
    int lane  = threadIdx.x & 31;
    int half  = lane >> 4;
    int l16   = lane & 15;

    int qb = wave & 127;                  // 16-query block
    int bh = wave >> 7;                   // 0..63

    const unsigned short* Q = Qb + ((size_t)bh * S_DIM + qb * 16) * D_DIM;
    const unsigned short* K = Kb + (size_t)bh * S_DIM * D_DIM;
    const unsigned short* V = VT + (size_t)bh * D_DIM * S_DIM;

    // Q A-fragments for K-dim 0..31 and 32..63
    const unsigned short* qrow = Q + (size_t)l16 * D_DIM;
    v16bf qf[2];
#pragma unroll
    for (int t = 0; t < 2; ++t) {
        v8bf lo = *(const v8bf*)(qrow + t * 32 + half * 16);
        v8bf hi = *(const v8bf*)(qrow + t * 32 + half * 16 + 8);
        qf[t] = cat8(lo, hi);
    }

    float mrow[8], lrow[8];
#pragma unroll
    for (int r = 0; r < 8; ++r) { mrow[r] = -3.0e38f; lrow[r] = 0.0f; }
    v8f o[4] = {};
    unsigned short* pl = &pbuf[wslot][0];

    for (int n0 = 0; n0 < S_DIM; n0 += 32) {
        // scores for 32 keys: two 16-wide C tiles
        v8f s0 = {}, s1 = {};
#pragma unroll
        for (int kk = 0; kk < 2; ++kk) {           // K-dim over D
            const unsigned short* kb = K + (size_t)(n0 + l16) * D_DIM + kk * 32 + half * 16;
            v16bf b0 = *(const v16bf*)kb;                      // keys n0..n0+15
            v16bf b1 = *(const v16bf*)(kb + 16 * D_DIM);       // keys n0+16..n0+31
            s0 = wmma_bf16(qf[kk], b0, s0);
            s1 = wmma_bf16(qf[kk], b1, s1);
        }
        // online softmax over this 32-key slab
#pragma unroll
        for (int r = 0; r < 8; ++r) {
            float t0 = s0[r], t1 = s1[r];
            float mx = fmaxf(t0, t1);
#pragma unroll
            for (int d = 8; d >= 1; d >>= 1) mx = fmaxf(mx, __shfl_xor(mx, d, 32));
            float mnew = fmaxf(mrow[r], mx);
            float corr = __expf(mrow[r] - mnew);
            float p0 = __expf(t0 - mnew);
            float p1 = __expf(t1 - mnew);
            float rs = p0 + p1;
#pragma unroll
            for (int d = 8; d >= 1; d >>= 1) rs += __shfl_xor(rs, d, 32);
            lrow[r] = lrow[r] * corr + rs;
            mrow[r] = mnew;
            o[0][r] *= corr; o[1][r] *= corr; o[2][r] *= corr; o[3][r] *= corr;
            // C-layout -> LDS row-major P[16][32] (bf16)
            int row = half * 8 + r;
            pl[row * 32 + l16]      = f2bf_rne(p0);
            pl[row * 32 + 16 + l16] = f2bf_rne(p1);
        }
        // LDS -> A-fragment (same-wave DS ops are in order; compiler waits DScnt on use)
        {
            const unsigned short* pr = pl + l16 * 32;
            v8bf lo = *(const v8bf*)(pr + half * 16);
            v8bf hi = *(const v8bf*)(pr + half * 16 + 8);
            v16bf pa = cat8(lo, hi);
#pragma unroll
            for (int j = 0; j < 4; ++j) {
                const unsigned short* vb = V + (size_t)(j * 16 + l16) * S_DIM + n0 + half * 16;
                o[j] = wmma_bf16(pa, *(const v16bf*)vb, o[j]);
            }
        }
    }

    // normalize and write O[B,S,E]
    int b = bh >> 4, h = bh & 15;
#pragma unroll
    for (int j = 0; j < 4; ++j) {
        int ecol = h * 64 + j * 16 + l16;
#pragma unroll
        for (int r = 0; r < 8; ++r) {
            int s = qb * 16 + half * 8 + r;
            Ob[((size_t)b * S_DIM + s) * E_DIM + ecol] = f2bf_rne(o[j][r] / lrow[r]);
        }
    }
}

// ---------- phase 3: output projection + bias + residual (32x64 per wave) ----------
__global__ void __launch_bounds__(256) oproj_kernel(
    const unsigned short* __restrict__ Ob,
    const unsigned short* __restrict__ woT,
    const float* __restrict__ bo,
    const float* __restrict__ x,
    float* __restrict__ tmp)
{
    int wave  = blockIdx.x * 8 + (threadIdx.x >> 5);
    int lane  = threadIdx.x & 31;
    int half  = lane >> 4;
    int l16   = lane & 15;
    int ng    = wave & 15;
    int mtile = wave >> 4;                 // 0..255
    int m0    = mtile * 32;
    const unsigned short* arow0 = Ob + (size_t)(m0 + l16) * E_DIM;
    const unsigned short* arow1 = Ob + (size_t)(m0 + 16 + l16) * E_DIM;
    const unsigned short* brow0 = woT + (size_t)(ng * 64 + l16) * E_DIM + half * 16;

    v8f acc[2][4] = {};
    for (int k0 = 0; k0 < E_DIM; k0 += 32) {
        v16bf a0 = cat8(*(const v8bf*)(arow0 + k0 + half * 16),
                        *(const v8bf*)(arow0 + k0 + half * 16 + 8));
        v16bf a1 = cat8(*(const v8bf*)(arow1 + k0 + half * 16),
                        *(const v8bf*)(arow1 + k0 + half * 16 + 8));
        __builtin_prefetch(arow0 + k0 + 32 + half * 8, 0, 1);
        __builtin_prefetch(arow1 + k0 + 32 + half * 8, 0, 1);
#pragma unroll
        for (int j = 0; j < 4; ++j) {
            v16bf b = *(const v16bf*)(brow0 + (size_t)(j * 16) * E_DIM + k0);
            acc[0][j] = wmma_bf16(a0, b, acc[0][j]);
            acc[1][j] = wmma_bf16(a1, b, acc[1][j]);
        }
    }
#pragma unroll
    for (int j = 0; j < 4; ++j) {
        int e = ng * 64 + j * 16 + l16;
        float bias = bo[e];
#pragma unroll
        for (int g = 0; g < 2; ++g) {
#pragma unroll
            for (int r = 0; r < 8; ++r) {
                int row = m0 + g * 16 + half * 8 + r;
                size_t idx = (size_t)row * E_DIM + e;
                tmp[idx] = acc[g][j][r] + bias + x[idx];
            }
        }
    }
}

// ---------- phase 4: LayerNorm (block per row) ----------
__global__ void __launch_bounds__(256) layernorm_kernel(
    const float* __restrict__ tmp, const float* __restrict__ gamma,
    const float* __restrict__ beta, float* __restrict__ out)
{
    __shared__ float red[2][8];
    int row = blockIdx.x;
    const float* xr = tmp + (size_t)row * E_DIM;
    int tid = threadIdx.x;
    float v[4];
    float sum = 0.f, sq = 0.f;
#pragma unroll
    for (int i = 0; i < 4; ++i) {
        v[i] = xr[tid + i * 256];
        sum += v[i]; sq += v[i] * v[i];
    }
#pragma unroll
    for (int d = 16; d >= 1; d >>= 1) {
        sum += __shfl_xor(sum, d, 32);
        sq  += __shfl_xor(sq,  d, 32);
    }
    int w = tid >> 5;
    if ((tid & 31) == 0) { red[0][w] = sum; red[1][w] = sq; }
    __syncthreads();
    sum = 0.f; sq = 0.f;
#pragma unroll
    for (int i = 0; i < 8; ++i) { sum += red[0][i]; sq += red[1][i]; }
    float mean = sum * (1.0f / E_DIM);
    float var  = sq * (1.0f / E_DIM) - mean * mean;
    float inv  = rsqrtf(var + 1e-6f);
    float* orow = out + (size_t)row * E_DIM;
#pragma unroll
    for (int i = 0; i < 4; ++i) {
        int c = tid + i * 256;
        orow[c] = (v[i] - mean) * inv * gamma[c] + beta[c];
    }
}

// ---------- launch ----------
extern "C" void kernel_launch(void* const* d_in, const int* in_sizes, int n_in,
                              void* d_out, int out_size, void* d_ws, size_t ws_size,
                              hipStream_t stream) {
    const float* x     = (const float*)d_in[0];
    const float* wq    = (const float*)d_in[1];
    const float* bq    = (const float*)d_in[2];
    const float* wk    = (const float*)d_in[3];
    const float* bk    = (const float*)d_in[4];
    const float* wv    = (const float*)d_in[5];
    const float* bv    = (const float*)d_in[6];
    const float* wo    = (const float*)d_in[7];
    const float* bo    = (const float*)d_in[8];
    const float* gamma = (const float*)d_in[9];
    const float* beta  = (const float*)d_in[10];
    float* out = (float*)d_out;

    // workspace layout (bf16 stored as ushort)
    char* ws = (char*)d_ws;
    size_t off = 0;
    unsigned short* xb = (unsigned short*)(ws + off); off += (size_t)M_ROWS * E_DIM * 2;       // 16 MB
    unsigned short* wt = (unsigned short*)(ws + off); off += (size_t)4 * E_DIM * E_DIM * 2;    //  8 MB
    unsigned short* Qb = (unsigned short*)(ws + off); off += (size_t)M_ROWS * E_DIM * 2;       // 16 MB
    unsigned short* Kb = (unsigned short*)(ws + off); off += (size_t)M_ROWS * E_DIM * 2;       // 16 MB
    unsigned short* VT = (unsigned short*)(ws + off); off += (size_t)M_ROWS * E_DIM * 2;       // 16 MB
    unsigned short* Ob = (unsigned short*)(ws + off); off += (size_t)M_ROWS * E_DIM * 2;       // 16 MB
    float* tmp = (float*)(ws + off);                  off += (size_t)M_ROWS * E_DIM * 4;       // 32 MB
    unsigned short* woT = wt + ((size_t)3 << 20);

    int nx = M_ROWS * E_DIM;          // 8388608
    convert_x_kernel<<<nx / 256, 256, 0, stream>>>(x, xb, nx);
    convert_w_kernel<<<(4 * E_DIM * E_DIM) / 256, 256, 0, stream>>>(wq, wk, wv, wo, wt);

    // 3 mats x 256 mtiles(32 rows) x 16 ngroups = 12288 waves; 8 waves/block
    qkv_gemm_kernel<<<1536, 256, 0, stream>>>(xb, wt, bq, bk, bv, Qb, Kb, VT);

    // 64 (b,h) x 128 qblocks = 8192 waves; 8 waves/block
    attention_kernel<<<1024, 256, 0, stream>>>(Qb, Kb, VT, Ob);

    // 256 mtiles(32 rows) x 16 ngroups = 4096 waves; 8 waves/block
    oproj_kernel<<<512, 256, 0, stream>>>(Ob, woT, bo, x, tmp);

    layernorm_kernel<<<M_ROWS, 256, 0, stream>>>(tmp, gamma, beta, out);
}
